// PreservedBidirectionalAttention_38044820308307
// MI455X (gfx1250) — compile-verified, matrix operand order checked by reading, not run on hardware
//
#include <hip/hip_runtime.h>
#include <hip/hip_bf16.h>
#include <math.h>

// ---------------- types ----------------
typedef __attribute__((ext_vector_type(16))) __bf16 v16bf;
typedef __attribute__((ext_vector_type(8)))  float  v8f;
typedef __attribute__((ext_vector_type(4)))  unsigned int v4u;
typedef __attribute__((ext_vector_type(8)))  int    v8i;
typedef __attribute__((ext_vector_type(4)))  int    v4i;

#define SEQ   2048
#define ATT   512
#define DV    512
#define NB    8

// fp32 -> bf16 round-to-nearest-even
__device__ __forceinline__ unsigned int f2bf(float f) {
    unsigned int u = __builtin_bit_cast(unsigned int, f);
    u += 0x7FFFu + ((u >> 16) & 1u);
    return u >> 16;
}
__device__ __forceinline__ unsigned int pack_bf2(float lo, float hi) {
    return f2bf(lo) | (f2bf(hi) << 16);
}

// ---------------- TDM: async 2D bf16 tile load (Global -> LDS) ----------------
// Tensor DMA Descriptor (CDNA5 ISA): 128 rows x 32 bf16,
// row stride `stride_elems`, landing contiguously in LDS (64 B/row -> matches
// the [128][16] u32 tile layout).
__device__ __forceinline__ void tdm_load_tile_bf16(unsigned lds_off,
                                                   const unsigned short* gsrc,
                                                   unsigned stride_elems) {
    unsigned long long ga = (unsigned long long)(uintptr_t)gsrc;
    v4u g0;
    g0.x = 1u;                                            // count=1, no gather
    g0.y = lds_off;                                       // lds_addr (bytes)
    g0.z = (unsigned)(ga & 0xFFFFFFFFu);                  // global_addr[31:0]
    g0.w = (unsigned)((ga >> 32) & 0x01FFFFFFu)           // global_addr[56:32]
         | (2u << 30);                                    // type = 2 ("image")
    v8i g1;
    g1[0] = (int)(1u << 16);                              // data_size=1 (2 bytes)
    g1[1] = (int)((ATT & 0xFFFFu) << 16);                 // tensor_dim0 lo16
    g1[2] = (int)(((ATT >> 16) & 0xFFFFu)                 // tensor_dim0 hi16
         | ((SEQ & 0xFFFFu) << 16));                      // tensor_dim1 lo16
    g1[3] = (int)(((SEQ >> 16) & 0xFFFFu)                 // tensor_dim1 hi16
         | (32u << 16));                                  // tile_dim0 = 32 elems
    g1[4] = (int)128u;                                    // tile_dim1=128, tile_dim2=0
    g1[5] = (int)stride_elems;                            // tensor_dim0_stride[31:0]
    g1[6] = 0;                                            // stride hi / dim1_stride
    g1[7] = 0;
    v4i z4 = (v4i)0;
#if __clang_major__ >= 23
    v8i z8 = (v8i)0;
    __builtin_amdgcn_tensor_load_to_lds(g0, g1, z4, z4, z8, 0);
#else
    __builtin_amdgcn_tensor_load_to_lds(g0, g1, z4, z4, 0);
#endif
}

// ---------------- WMMA core ----------------
// LDS tiles: [128 rows][16 u32] = row-major bf16 pairs over K (BK=32).
// A rows = M rows; B rows = N columns (N-major), so both fragment loads are
// identical 16B-run gathers matching the ISA 16-bit A 16x32 lane layout.
__device__ __forceinline__ v16bf frag_ld(const unsigned int (*T)[16], int rowBase, int lane) {
    const int row = rowBase + (lane & 15);
    const int bp  = (lane >> 4) << 2;
    union { v4u u[2]; v16bf v; } f;
    f.u[0] = *(const v4u*)&T[row][bp];
    f.u[1] = *(const v4u*)&T[row][8 + bp];
    return f.v;
}

__device__ __forceinline__ void wave_mma_step(const unsigned int (*Alds)[16],
                                              const unsigned int (*Blds)[16],
                                              int wm, int wn, int lane, v8f acc[2][4]) {
    v16bf a0 = frag_ld(Alds, wm * 32 +  0, lane);
    v16bf a1 = frag_ld(Alds, wm * 32 + 16, lane);
    v16bf b0 = frag_ld(Blds, wn * 64 +  0, lane);
    v16bf b1 = frag_ld(Blds, wn * 64 + 16, lane);
    v16bf b2 = frag_ld(Blds, wn * 64 + 32, lane);
    v16bf b3 = frag_ld(Blds, wn * 64 + 48, lane);
    acc[0][0] = __builtin_amdgcn_wmma_f32_16x16x32_bf16(false, a0, false, b0, (short)0, acc[0][0], false, false);
    acc[0][1] = __builtin_amdgcn_wmma_f32_16x16x32_bf16(false, a0, false, b1, (short)0, acc[0][1], false, false);
    acc[0][2] = __builtin_amdgcn_wmma_f32_16x16x32_bf16(false, a0, false, b2, (short)0, acc[0][2], false, false);
    acc[0][3] = __builtin_amdgcn_wmma_f32_16x16x32_bf16(false, a0, false, b3, (short)0, acc[0][3], false, false);
    acc[1][0] = __builtin_amdgcn_wmma_f32_16x16x32_bf16(false, a1, false, b0, (short)0, acc[1][0], false, false);
    acc[1][1] = __builtin_amdgcn_wmma_f32_16x16x32_bf16(false, a1, false, b1, (short)0, acc[1][1], false, false);
    acc[1][2] = __builtin_amdgcn_wmma_f32_16x16x32_bf16(false, a1, false, b2, (short)0, acc[1][2], false, false);
    acc[1][3] = __builtin_amdgcn_wmma_f32_16x16x32_bf16(false, a1, false, b3, (short)0, acc[1][3], false, false);
}

// ---------------- Kernel 1: projection GEMM  Out_bf16[M x 512] = concat(A0,A1) @ W + bias ----
// grid: (M/128, 512/128), block 256 (8 waves)
__global__ __launch_bounds__(256)
void proj_gemm_kernel(const float* __restrict__ A0, const float* __restrict__ A1,
                      int k0, int Ktot,
                      const float* __restrict__ W, const float* __restrict__ bias,
                      unsigned short* __restrict__ Out) {
    __shared__ unsigned int Alds[128][16];
    __shared__ unsigned int Blds[128][16];

    const int t    = threadIdx.x;
    const int lane = t & 31;
    const int wave = t >> 5;
    const int wm   = wave & 3;       // 4 waves down (rows)
    const int wn   = wave >> 2;      // 2 waves across (cols)
    const int M0   = blockIdx.x * 128;
    const int N0   = blockIdx.y * 128;
    const int k1w  = Ktot - k0;      // width of A1

    v8f acc[2][4];
    #pragma unroll
    for (int i = 0; i < 2; i++)
        #pragma unroll
        for (int j = 0; j < 4; j++) acc[i][j] = (v8f)0.0f;

    for (int kk = 0; kk < Ktot; kk += 32) {
        __syncthreads();
        // --- stage A: concat boundary (k0) is a multiple of the 16-elem run, so
        //     each run comes entirely from one source -> contiguous b128 loads ---
        {
            const int r  = t >> 1;
            const int cb = (t & 1) * 16;
            const int row = M0 + r;
            const int gk  = kk + cb;
            const float* src = (gk < k0) ? &A0[(size_t)row * k0 + gk]
                                         : &A1[(size_t)row * k1w + (gk - k0)];
            float f[16];
            #pragma unroll
            for (int c = 0; c < 16; c++) f[c] = src[c];
            #pragma unroll
            for (int j = 0; j < 8; j++)
                Alds[r][(cb >> 1) + j] = pack_bf2(f[2 * j], f[2 * j + 1]);
        }
        // --- stage B transposed: Blds[n][k] = W[kk+k][N0+n] ---
        {
            const int k  = (t >> 4) * 2;           // 0..30 even
            const int nb = (t & 15) * 8;           // 0..120
            const float* w0 = &W[(size_t)(kk + k) * ATT + N0 + nb];
            const float* w1 = &W[(size_t)(kk + k + 1) * ATT + N0 + nb];
            float f0[8], f1[8];
            #pragma unroll
            for (int i = 0; i < 8; i++) { f0[i] = w0[i]; f1[i] = w1[i]; }
            #pragma unroll
            for (int i = 0; i < 8; i++)
                Blds[nb + i][k >> 1] = pack_bf2(f0[i], f1[i]);
        }
        __syncthreads();
        wave_mma_step(Alds, Blds, wm, wn, lane, acc);
    }

    // --- store with fused bias, fp32 -> bf16 ---
    const int half = lane >> 4;
    #pragma unroll
    for (int mi = 0; mi < 2; mi++)
        #pragma unroll
        for (int ni = 0; ni < 4; ni++)
            #pragma unroll
            for (int r = 0; r < 8; r++) {
                const int grow = M0 + wm * 32 + mi * 16 + half * 8 + r;
                const int gcol = N0 + wn * 64 + ni * 16 + (lane & 15);
                const float v  = acc[mi][ni][r] + bias[gcol];
                Out[(size_t)grow * ATT + gcol] = (unsigned short)f2bf(v);
            }
}

// ---------------- Kernel 2: score GEMM  Out[b,i,j] = X[b,i,:]·Y[b,j,:], XOR-masked -inf ----
// Double-buffered TDM pipeline: tile k+1 is in flight (tensorcnt<=2) while the
// WGP runs WMMA on tile k. In-order TDM completion makes the partial wait safe.
// grid: (16,16,NB), block 256
__global__ __launch_bounds__(256)
void score_gemm_kernel(const unsigned short* __restrict__ X,
                       const unsigned short* __restrict__ Y,
                       float* __restrict__ Out,
                       const long long* __restrict__ lrow,
                       const long long* __restrict__ lcol) {
    __shared__ unsigned int Alds[2][128][16];
    __shared__ unsigned int Blds[2][128][16];

    const int t    = threadIdx.x;
    const int lane = t & 31;
    const int wave = t >> 5;
    const int wm   = wave & 3;
    const int wn   = wave >> 2;
    const int b    = blockIdx.z;
    const int i0   = blockIdx.x * 128;
    const int j0   = blockIdx.y * 128;

    const unsigned short* Xb = X + (size_t)b * SEQ * ATT;
    const unsigned short* Yb = Y + (size_t)b * SEQ * ATT;
    float* Ob = Out + (size_t)b * SEQ * SEQ;

    const unsigned ldsA0 = (unsigned)(uintptr_t)(&Alds[0][0][0]); // low 32 bits = LDS offset
    const unsigned ldsA1 = (unsigned)(uintptr_t)(&Alds[1][0][0]);
    const unsigned ldsB0 = (unsigned)(uintptr_t)(&Blds[0][0][0]);
    const unsigned ldsB1 = (unsigned)(uintptr_t)(&Blds[1][0][0]);

    v8f acc[2][4];
    #pragma unroll
    for (int i = 0; i < 2; i++)
        #pragma unroll
        for (int j = 0; j < 4; j++) acc[i][j] = (v8f)0.0f;

    if (t == 0) {                        // TDM ignores EXEC; issue once per block
        tdm_load_tile_bf16(ldsA0, Xb + (size_t)i0 * ATT, ATT);
        tdm_load_tile_bf16(ldsB0, Yb + (size_t)j0 * ATT, ATT);
    }

    int s = 0;
    for (int kk = 0; kk < ATT; kk += 32, s ^= 1) {
        __syncthreads();                 // all waves done reading buffer s^1
        if (t == 0) {
            if (kk + 32 < ATT) {         // issue next tile pair into buffer s^1
                tdm_load_tile_bf16(s ? ldsA0 : ldsA1, Xb + (size_t)i0 * ATT + kk + 32, ATT);
                tdm_load_tile_bf16(s ? ldsB0 : ldsB1, Yb + (size_t)j0 * ATT + kk + 32, ATT);
                __builtin_amdgcn_s_wait_tensorcnt(2);  // buffer s complete; next pair in flight
            } else {
                __builtin_amdgcn_s_wait_tensorcnt(0);
            }
        }
        __syncthreads();                 // publish buffer s to all waves
        wave_mma_step(Alds[s], Blds[s], wm, wn, lane, acc);
    }

    const int Lr = (int)lrow[b];
    const int Lc = (int)lcol[b];
    const float ninf = -__builtin_inff();
    const int half = lane >> 4;
    #pragma unroll
    for (int mi = 0; mi < 2; mi++)
        #pragma unroll
        for (int ni = 0; ni < 4; ni++)
            #pragma unroll
            for (int rr = 0; rr < 8; rr++) {
                const int grow = i0 + wm * 32 + mi * 16 + half * 8 + rr;
                const int gcol = j0 + wn * 64 + ni * 16 + (lane & 15);
                const bool m = ((gcol >= Lc) != (grow >= Lr));   // exactly-one => XOR
                Ob[(size_t)grow * SEQ + gcol] = m ? ninf : acc[mi][ni][rr];
            }
}

// ---------------- Kernel 3: in-place row softmax over 2048 ----
// grid: (SEQ, NB), block 256; each thread handles 8 strided elements.
__global__ __launch_bounds__(256)
void softmax_kernel(float* __restrict__ P) {
    __shared__ float redA[8];
    __shared__ float redB[8];
    const int t = threadIdx.x, lane = t & 31, wave = t >> 5;
    float* row = P + ((size_t)blockIdx.y * SEQ + blockIdx.x) * SEQ;

    float f[8];
    #pragma unroll
    for (int j = 0; j < 8; j++) f[j] = row[t + j * 256];

    float m = f[0];
    #pragma unroll
    for (int j = 1; j < 8; j++) m = fmaxf(m, f[j]);
    #pragma unroll
    for (int off = 16; off > 0; off >>= 1) m = fmaxf(m, __shfl_down(m, off, 32));
    if (lane == 0) redA[wave] = m;
    __syncthreads();
    float rmax = redA[0];
    #pragma unroll
    for (int w = 1; w < 8; w++) rmax = fmaxf(rmax, redA[w]);

    float s = 0.0f;
    #pragma unroll
    for (int j = 0; j < 8; j++) { f[j] = __expf(f[j] - rmax); s += f[j]; }
    #pragma unroll
    for (int off = 16; off > 0; off >>= 1) s += __shfl_down(s, off, 32);
    if (lane == 0) redB[wave] = s;
    __syncthreads();
    float rsum = 0.0f;
    #pragma unroll
    for (int w = 0; w < 8; w++) rsum += redB[w];
    const float inv = 1.0f / rsum;

    #pragma unroll
    for (int j = 0; j < 8; j++) row[t + j * 256] = f[j] * inv;
}

// ---------------- Kernel 4: output GEMM  O[b] = Wm[b] @ V[b]  (fp32 in, bf16 WMMA) ----
// grid: (16, 4, NB), block 256
__global__ __launch_bounds__(256)
void out_gemm_kernel(const float* __restrict__ Wm, const float* __restrict__ V,
                     float* __restrict__ O) {
    __shared__ unsigned int Alds[128][16];
    __shared__ unsigned int Blds[128][16];

    const int t    = threadIdx.x;
    const int lane = t & 31;
    const int wave = t >> 5;
    const int wm   = wave & 3;
    const int wn   = wave >> 2;
    const int b    = blockIdx.z;
    const int i0   = blockIdx.x * 128;
    const int n0   = blockIdx.y * 128;

    const float* Wb = Wm + (size_t)b * SEQ * SEQ;
    const float* Vb = V  + (size_t)b * SEQ * DV;
    float* Ob = O + (size_t)b * SEQ * DV;

    v8f acc[2][4];
    #pragma unroll
    for (int i = 0; i < 2; i++)
        #pragma unroll
        for (int j = 0; j < 4; j++) acc[i][j] = (v8f)0.0f;

    for (int kk = 0; kk < SEQ; kk += 32) {
        __syncthreads();
        {   // stage A: fp32 w row -> bf16 pairs (134 MB stream; prefetch next tile)
            const int r  = t >> 1;
            const int cb = (t & 1) * 16;
            const float* src = &Wb[(size_t)(i0 + r) * SEQ + kk + cb];
            if (kk + 32 < SEQ) __builtin_prefetch(src + 32, 0, 0);
            float f[16];
            #pragma unroll
            for (int c = 0; c < 16; c++) f[c] = src[c];
            #pragma unroll
            for (int j = 0; j < 8; j++)
                Alds[r][(cb >> 1) + j] = pack_bf2(f[2 * j], f[2 * j + 1]);
        }
        {   // stage B transposed: Blds[n][k] = V[kk+k][n0+n]
            const int k  = (t >> 4) * 2;
            const int nb = (t & 15) * 8;
            const float* v0 = &Vb[(size_t)(kk + k) * DV + n0 + nb];
            const float* v1 = &Vb[(size_t)(kk + k + 1) * DV + n0 + nb];
            float f0[8], f1[8];
            #pragma unroll
            for (int i = 0; i < 8; i++) { f0[i] = v0[i]; f1[i] = v1[i]; }
            #pragma unroll
            for (int i = 0; i < 8; i++)
                Blds[nb + i][k >> 1] = pack_bf2(f0[i], f1[i]);
        }
        __syncthreads();
        wave_mma_step(Alds, Blds, wm, wn, lane, acc);
    }

    const int half = lane >> 4;
    #pragma unroll
    for (int mi = 0; mi < 2; mi++)
        #pragma unroll
        for (int ni = 0; ni < 4; ni++)
            #pragma unroll
            for (int r = 0; r < 8; r++) {
                const int grow = i0 + wm * 32 + mi * 16 + half * 8 + r;
                const int gcol = n0 + wn * 64 + ni * 16 + (lane & 15);
                Ob[(size_t)grow * DV + gcol] = acc[mi][ni][r];
            }
}

// ---------------- launch ----------------
extern "C" void kernel_launch(void* const* d_in, const int* in_sizes, int n_in,
                              void* d_out, int out_size, void* d_ws, size_t ws_size,
                              hipStream_t stream) {
    (void)in_sizes; (void)n_in; (void)out_size; (void)ws_size;

    const float* k1   = (const float*)d_in[0];
    const float* k2   = (const float*)d_in[1];
    const float* pk1s = (const float*)d_in[2];
    const float* pk2s = (const float*)d_in[3];
    const float* v1   = (const float*)d_in[4];
    const float* v2   = (const float*)d_in[5];
    const float* W_k1 = (const float*)d_in[6];
    const float* b_k1 = (const float*)d_in[7];
    const float* W_k2 = (const float*)d_in[8];
    const float* b_k2 = (const float*)d_in[9];
    const float* W_p1 = (const float*)d_in[10];
    const float* b_p1 = (const float*)d_in[11];
    const float* W_p2 = (const float*)d_in[12];
    const float* b_p2 = (const float*)d_in[13];
    const long long* l1 = (const long long*)d_in[14];
    const long long* l2 = (const long long*)d_in[15];

    // Output layout (floats): o1 | o2 | w1 | w2
    float* out = (float*)d_out;
    const size_t oSz = (size_t)NB * SEQ * DV;      // 8,388,608
    const size_t wSz = (size_t)NB * SEQ * SEQ;     // 33,554,432
    float* o1 = out;
    float* o2 = out + oSz;
    float* w1 = out + 2 * oSz;
    float* w2 = out + 2 * oSz + wSz;

    // workspace: two bf16 projection buffers [NB*SEQ*ATT] each (16.8 MB apiece)
    unsigned short* buf0 = (unsigned short*)d_ws;
    unsigned short* buf1 = buf0 + (size_t)NB * SEQ * ATT;

    const dim3 blk(256);
    const dim3 gProj(128, 4);          // 16384/128 x 512/128
    const dim3 gScore(16, 16, NB);     // 2048/128 x 2048/128 x B
    const dim3 gSoft(SEQ, NB);
    const dim3 gOut(16, 4, NB);        // 2048/128 x 512/128 x B

    // ---- side 1: w1 = softmax_s( pk2 · k1p^T ), o1 = w1 @ v1 ----
    proj_gemm_kernel<<<gProj, blk, 0, stream>>>(k1, nullptr, ATT, ATT, W_k1, b_k1, buf0);        // k1p
    proj_gemm_kernel<<<gProj, blk, 0, stream>>>(k2, pk2s, ATT, ATT + 256, W_p2, b_p2, buf1);     // pk2
    // rows t (S2) from pk2, cols s (S1) from k1p; mask: (s>=l1) XOR (t>=l2)
    score_gemm_kernel<<<gScore, blk, 0, stream>>>(buf1, buf0, w1, l2, l1);
    softmax_kernel<<<gSoft, blk, 0, stream>>>(w1);
    out_gemm_kernel<<<gOut, blk, 0, stream>>>(w1, v1, o1);

    // ---- side 2: w2 = softmax_t( pk1 · k2p^T ), o2 = w2 @ v2 ----
    proj_gemm_kernel<<<gProj, blk, 0, stream>>>(k1, pk1s, ATT, ATT + 256, W_p1, b_p1, buf0);     // pk1
    proj_gemm_kernel<<<gProj, blk, 0, stream>>>(k2, nullptr, ATT, ATT, W_k2, b_k2, buf1);        // k2p
    // rows s (S1) from pk1, cols t (S2) from k2p; mask: (s>=l1) XOR (t>=l2)
    score_gemm_kernel<<<gScore, blk, 0, stream>>>(buf0, buf1, w2, l1, l2);
    softmax_kernel<<<gSoft, blk, 0, stream>>>(w2);
    out_gemm_kernel<<<gOut, blk, 0, stream>>>(w2, v2, o2);
}
